// Policy_36223754174568
// MI455X (gfx1250) — matrix operand order repaired and checked by reading, not compiled, not run on hardware
//
#include <hip/hip_runtime.h>
#include <hip/hip_bf16.h>
#include <stdint.h>

#define BB   64
#define SS   256
#define HHID 128
#define TOK  (BB*SS)   // 16384 tokens

typedef __attribute__((ext_vector_type(16))) __bf16 bf16x16;
typedef __attribute__((ext_vector_type(8)))  float  f32x8;

union ABfrag { bf16x16 v; unsigned int u[8]; };
union CDfrag { f32x8   v; float        f[8]; };

__device__ __forceinline__ unsigned short f2bf(float x){
  unsigned int u = __float_as_uint(x);
  unsigned int r = u + 0x7FFFu + ((u >> 16) & 1u);   // round-to-nearest-even
  return (unsigned short)(r >> 16);
}
__device__ __forceinline__ unsigned int packbf2(float a, float b){
  return (unsigned int)f2bf(a) | ((unsigned int)f2bf(b) << 16);
}
// order-preserving float<->uint map (for deterministic atomic max-pool)
__device__ __forceinline__ unsigned int ordf(float x){
  unsigned int u = __float_as_uint(x);
  return (u & 0x80000000u) ? ~u : (u | 0x80000000u);
}
__device__ __forceinline__ float unordf(unsigned int u){
  u = (u & 0x80000000u) ? (u & 0x7FFFFFFFu) : ~u;
  return __uint_as_float(u);
}
__device__ __forceinline__ float sigm(float x){ return 1.0f/(1.0f + __expf(-x)); }

// --- WMMA fragment builders (contiguous k = 16*(lane/16) + s, same in A and B) ---------------
// A (16x32 bf16): lanes0-15 M=lane, u[0..7]=K 0..15 pairs;
//                 lanes16-31 same M, K offset +16.
__device__ __forceinline__ ABfrag load_a_frag(const unsigned short* lds, int row0,
                                              int ldk, int kbase, int lane){
  ABfrag a;
  const int m  = lane & 15;
  const int ko = (lane < 16) ? 0 : 16;
  const unsigned short* p = lds + (size_t)(row0 + m) * ldk + kbase;
  #pragma unroll
  for (int j = 0; j < 4; ++j) a.u[j]   = *(const unsigned int*)(p + ko + 2*j);
  #pragma unroll
  for (int j = 0; j < 4; ++j) a.u[4+j] = *(const unsigned int*)(p + 8 + ko + 2*j);
  return a;
}
// B (32x16 bf16): lanes0-15 N=lane, K=2j,2j+1 per VGPR; lanes16-31 K offset +16.
// LDS staged n-major ([N][K]) so K-pairs are contiguous u32s.
__device__ __forceinline__ ABfrag load_b_frag(const unsigned short* lds, int n0,
                                              int ldk, int kbase, int lane){
  ABfrag b;
  const int n  = lane & 15;
  const int ko = (lane < 16) ? 0 : 16;
  const unsigned short* p = lds + (size_t)(n0 + n) * ldk + kbase + ko;
  #pragma unroll
  for (int j = 0; j < 8; ++j) b.u[j] = *(const unsigned int*)(p + 2*j);
  return b;
}
#define WMMA_BF16(acc, a, b) \
  (acc).v = __builtin_amdgcn_wmma_f32_16x16x32_bf16(false, (a).v, false, (b).v, \
                                                    (short)0, (acc).v, false, false)

// =============================================================================
// Generic GEMM: C[M,N] = A[M,K](f32) * B^T + bias, B given as W[N,K] f32
// (transB=1 reads W as [K,N]). f32->bf16 conversion during vectorized LDS
// staging (float4 -> global_load_b128; packed bf16 dword stores).
// ALIGNMENT CONTRACT (all call sites satisfy it): M%128==0, K%32==0, N%16==0,
// lda%4==0, row pointers 16B aligned.
// Block: 256 thr (8 waves); block tile 128(M) x 16(N); wave tile 16x16.
// =============================================================================
__global__ __launch_bounds__(256) void gemm_wmma_kernel(
    const float* __restrict__ A, const float* __restrict__ Bw,
    const float* __restrict__ bias, float* __restrict__ C,
    int M, int N, int K, int lda, int ldb, int ldc, int relu, int transB)
{
  __shared__ unsigned short As[128*32];
  __shared__ unsigned short Bs[16*32];
  const int tid  = threadIdx.x;
  const int lane = tid & 31;
  const int wave = tid >> 5;
  const int m0 = blockIdx.x * 128;
  const int n0 = blockIdx.y * 16;

  CDfrag acc;
  #pragma unroll
  for (int i = 0; i < 8; ++i) acc.f[i] = 0.0f;

  for (int k0 = 0; k0 < K; k0 += 32) {
    // ---- stage A tile 128x32 (vectorized, unguarded) ----
    {
      int e = tid * 4;                     // 4096 bf16 per tile
      #pragma unroll
      for (int it = 0; it < 4; ++it, e += 1024) {
        int r = e >> 5, c = e & 31;
        const float* src = A + (size_t)(m0 + r)*lda + k0 + c;
        if (k0 + 32 < K) __builtin_prefetch(src + 32, 0, 1);  // next k-tile -> global_prefetch
        float4 v = *(const float4*)src;
        *(unsigned int*)(As + e)     = packbf2(v.x, v.y);
        *(unsigned int*)(As + e + 2) = packbf2(v.z, v.w);
      }
    }
    // ---- stage B tile (n-major 16x32) ----
    if (!transB) {
      if (tid < 128) {
        int e = tid * 4;                   // 512 bf16 per tile
        int n = e >> 5, c = e & 31;
        float4 v = *(const float4*)(Bw + (size_t)(n0 + n)*ldb + k0 + c);
        *(unsigned int*)(Bs + e)     = packbf2(v.x, v.y);
        *(unsigned int*)(Bs + e + 2) = packbf2(v.z, v.w);
      }
    } else {
      for (int idx = tid; idx < 16*32; idx += 256) {
        int n = idx >> 5, c = idx & 31;
        Bs[idx] = f2bf(Bw[(size_t)(k0 + c)*ldb + (n0 + n)]);
      }
    }
    __syncthreads();
    ABfrag a = load_a_frag(As, wave*16, 32, 0, lane);
    ABfrag b = load_b_frag(Bs, 0, 32, 0, lane);
    WMMA_BF16(acc, a, b);
    __syncthreads();
  }

  const int n = n0 + (lane & 15);
  float bv = (bias != nullptr) ? bias[n] : 0.0f;
  #pragma unroll
  for (int i = 0; i < 8; ++i) {
    int row = m0 + wave*16 + i + ((lane < 16) ? 0 : 8);
    float v = acc.f[i] + bv;
    if (relu) v = fmaxf(v, 0.0f);
    C[(size_t)row*ldc + n] = v;
  }
}

// =============================================================================
// Fused unit embed: basic = relu(U[Mg,9] @ Wb^T + bb)  (K padded 9->32, WMMA)
//                   emb   = basic @ Wt^T + bt          (WMMA, K=128)
//                   pooled[token, g*128+n] = max over units  (atomic ord-max)
// Block: 256 thr (8 waves), 128 rows/block (Mg%128==0 for all groups).
// Static LDS = 64KB.
// =============================================================================
__global__ __launch_bounds__(256) void unit_embed_kernel(
    const float* __restrict__ U, const float* __restrict__ Wb, const float* __restrict__ bb,
    const float* __restrict__ Wt, const float* __restrict__ bt,
    unsigned int* __restrict__ pooledOrd, int Ng, int gIdx, int Mg)
{
  __shared__ unsigned short sBasic[128*128];   // 32KB (bf16 activations)
  __shared__ unsigned short sW[128*128];       // 32KB (stage1 scratch / stage2 Wt)
  const int tid = threadIdx.x, lane = tid & 31, wave = tid >> 5;
  const int m0 = blockIdx.x * 128;

  // stage1 staging: sW[0..4095] = zero-padded U tile, sW[4096..8191] = padded Wb
  // (K=9 is ragged -> keep the guarded scalar path here)
  for (int idx = tid; idx < 128*32; idx += 256) {
    int r = idx >> 5, c = idx & 31;
    float v = (c < 9) ? U[(size_t)(m0 + r)*9 + c] : 0.0f;
    sW[idx] = f2bf(v);
  }
  for (int idx = tid; idx < 128*32; idx += 256) {
    int n = idx >> 5, c = idx & 31;
    float v = (c < 9) ? Wb[(size_t)n*9 + c] : 0.0f;
    sW[4096 + idx] = f2bf(v);
  }
  __syncthreads();

  { // stage1: basic relu -> sBasic (bf16, row-major [128 rows][128 h])
    ABfrag a = load_a_frag(sW, wave*16, 32, 0, lane);
    const int nc = lane & 15;
    #pragma unroll
    for (int nt = 0; nt < 8; ++nt) {
      ABfrag b = load_b_frag(sW + 4096, nt*16, 32, 0, lane);
      CDfrag c;
      #pragma unroll
      for (int i = 0; i < 8; ++i) c.f[i] = 0.0f;
      WMMA_BF16(c, a, b);
      int n = nt*16 + nc;
      float bias = bb[n];
      #pragma unroll
      for (int i = 0; i < 8; ++i) {
        int row = wave*16 + i + ((lane < 16) ? 0 : 8);
        sBasic[(size_t)row*128 + n] = f2bf(fmaxf(c.f[i] + bias, 0.0f));
      }
    }
  }
  __syncthreads();

  // stage2: load Wt (128x128, dense) into sW -- vectorized float4 path
  {
    int e = tid * 4;                       // 16384 elements, 16 iterations
    #pragma unroll
    for (int it = 0; it < 16; ++it, e += 1024) {
      float4 v = *(const float4*)(Wt + e);
      *(unsigned int*)(sW + e)     = packbf2(v.x, v.y);
      *(unsigned int*)(sW + e + 2) = packbf2(v.z, v.w);
    }
  }
  __syncthreads();

  ABfrag a4[4];
  #pragma unroll
  for (int kt = 0; kt < 4; ++kt)
    a4[kt] = load_a_frag(sBasic, wave*16, 128, kt*32, lane);

  const int nc = lane & 15;
  #pragma unroll
  for (int nt = 0; nt < 8; ++nt) {
    CDfrag c;
    #pragma unroll
    for (int i = 0; i < 8; ++i) c.f[i] = 0.0f;
    #pragma unroll
    for (int kt = 0; kt < 4; ++kt) {
      ABfrag b = load_b_frag(sW, nt*16, 128, kt*32, lane);
      WMMA_BF16(c, a4[kt], b);
    }
    int n = nt*16 + nc;
    float bias = bt[n];
    #pragma unroll
    for (int i = 0; i < 8; ++i) {
      int row = wave*16 + i + ((lane < 16) ? 0 : 8);
      int gm = m0 + row;
      int token = gm / Ng;
      atomicMax(&pooledOrd[(size_t)token*512 + gIdx*128 + n], ordf(c.f[i] + bias));
    }
  }
}

// ============================ small helper kernels ===========================
__global__ void init_pooled_kernel(unsigned int* __restrict__ p){
  int i = blockIdx.x*256 + threadIdx.x;
  if (i < TOK*512) p[i] = 0u;
}
__global__ void decode_pooled_kernel(const unsigned int* __restrict__ p, float* __restrict__ x640){
  int i = blockIdx.x*256 + threadIdx.x;
  if (i >= TOK*512) return;
  int t = i >> 9, c = i & 511;
  x640[(size_t)t*640 + 128 + c] = unordf(p[i]);
}
__global__ void env_kernel(const float* __restrict__ env, const float* __restrict__ W,
                           const float* __restrict__ b, float* __restrict__ x640){
  int idx = blockIdx.x*256 + threadIdx.x;
  if (idx >= TOK*128) return;
  int t = idx >> 7, h = idx & 127;
  const float* e = env + (size_t)t*3;
  float v = b[h] + W[h*3+0]*e[0] + W[h*3+1]*e[1] + W[h*3+2]*e[2];
  x640[(size_t)t*640 + h] = fmaxf(v, 0.0f);
}
__global__ void bias_sum_kernel(const float* __restrict__ a, const float* __restrict__ b,
                                float* __restrict__ o){
  int i = blockIdx.x*256 + threadIdx.x;
  if (i < 512) o[i] = a[i] + b[i];
}

// =============================================================================
// Persistent LSTM: 4 blocks (batch tiles of 16), 8 waves (hidden tiles of 16).
// gates_x holds x@W_ih^T + (b_ih+b_hh) precomputed. Each wave keeps 16 W_hh
// B-fragments resident in VGPRs and its c-state tile in registers; h double-
// buffers as bf16 in LDS (the block's A operand for the next step).
// A block's output h rows are exactly the A rows it consumes next step, so the
// recurrence needs no inter-block synchronization.
// =============================================================================
__global__ __launch_bounds__(256) void lstm_kernel(
    const float* __restrict__ gates_x, const float* __restrict__ Whh,
    const float* __restrict__ h0, const float* __restrict__ c0,
    float* __restrict__ xout, float* __restrict__ hn, float* __restrict__ cn)
{
  __shared__ unsigned short hbuf[2][16*128];
  const int tid  = threadIdx.x, lane = tid & 31;
  const int ht   = tid >> 5;        // hidden tile 0..7
  const int mt   = blockIdx.x;      // batch tile 0..3
  const int nc   = lane & 15;
  const int hid  = ht*16 + nc;
  const int rsel = (lane < 16) ? 0 : 8;

  // resident W_hh fragments: [gate][ktile]
  ABfrag bw[4][4];
  #pragma unroll
  for (int g = 0; g < 4; ++g) {
    const float* wr = Whh + (size_t)(g*128 + hid) * 128;
    #pragma unroll
    for (int kt = 0; kt < 4; ++kt) {
      int kb = kt*32 + ((lane < 16) ? 0 : 16);
      #pragma unroll
      for (int j = 0; j < 8; ++j)
        bw[g][kt].u[j] = packbf2(wr[kb + 2*j], wr[kb + 2*j + 1]);
    }
  }

  CDfrag cfrag, hfrag;
  #pragma unroll
  for (int i = 0; i < 8; ++i) {
    int b = mt*16 + i + rsel;
    cfrag.f[i] = c0[(size_t)b*128 + hid];
    hfrag.f[i] = h0[(size_t)b*128 + hid];
  }
  for (int idx = tid; idx < 16*128; idx += 256) {
    int r = idx >> 7, c = idx & 127;
    hbuf[0][idx] = f2bf(h0[(size_t)(mt*16 + r)*128 + c]);
  }

  for (int t = 0; t < SS; ++t) {
    __syncthreads();
    const unsigned short* hb = hbuf[t & 1];
    ABfrag a4[4];
    #pragma unroll
    for (int kt = 0; kt < 4; ++kt)
      a4[kt] = load_a_frag(hb, 0, 128, kt*32, lane);

    CDfrag g4[4];
    #pragma unroll
    for (int g = 0; g < 4; ++g) {
      int n = g*128 + hid;
      #pragma unroll
      for (int i = 0; i < 8; ++i) {
        int b = mt*16 + i + rsel;
        g4[g].f[i] = gates_x[((size_t)b*SS + t)*512 + n];
      }
      #pragma unroll
      for (int kt = 0; kt < 4; ++kt)
        WMMA_BF16(g4[g], a4[kt], bw[g][kt]);
    }

    unsigned short* hnext = hbuf[(t+1) & 1];
    #pragma unroll
    for (int i = 0; i < 8; ++i) {
      int rl = i + rsel;
      int b  = mt*16 + rl;
      float iv = sigm(g4[0].f[i]);
      float fv = sigm(g4[1].f[i]);
      float gv = tanhf(g4[2].f[i]);
      float ov = sigm(g4[3].f[i]);
      float cc = fv * cfrag.f[i] + iv * gv;
      float hh = ov * tanhf(cc);
      cfrag.f[i] = cc;
      hfrag.f[i] = hh;
      xout[((size_t)b*SS + t)*128 + hid] = hh;
      hnext[rl*128 + hid] = f2bf(hh);
    }
  }

  #pragma unroll
  for (int i = 0; i < 8; ++i) {
    int b = mt*16 + i + rsel;
    hn[(size_t)b*128 + hid] = hfrag.f[i];
    cn[(size_t)b*128 + hid] = cfrag.f[i];
  }
}

// ======================= heads: enum/x/y softmax + att·b_g ===================
__global__ void heads_kernel(const float* __restrict__ x, const float* __restrict__ att,
    const float* __restrict__ We, const float* __restrict__ be,
    const float* __restrict__ Wx, const float* __restrict__ bx,
    const float* __restrict__ Wy, const float* __restrict__ by,
    const float* __restrict__ bt0, const float* __restrict__ bt1,
    const float* __restrict__ bt2, const float* __restrict__ bt3,
    float* __restrict__ out_enum, float* __restrict__ out_x, float* __restrict__ out_y,
    float* __restrict__ attb)
{
  int t = blockIdx.x*256 + threadIdx.x;
  if (t >= TOK) return;
  const float* xr = x + (size_t)t*128;
  float s[9];
  { // enum (3)
    for (int j = 0; j < 3; ++j) {
      float a = be[j]; const float* w = We + j*128;
      for (int h = 0; h < 128; ++h) a += w[h]*xr[h];
      s[j] = a;
    }
    float m = fmaxf(s[0], fmaxf(s[1], s[2]));
    float e0 = __expf(s[0]-m), e1 = __expf(s[1]-m), e2 = __expf(s[2]-m);
    float inv = 1.0f/(e0+e1+e2);
    out_enum[t*3+0]=e0*inv; out_enum[t*3+1]=e1*inv; out_enum[t*3+2]=e2*inv;
  }
  { // x (9)
    float m = -1e30f;
    for (int j = 0; j < 9; ++j) {
      float a = bx[j]; const float* w = Wx + j*128;
      for (int h = 0; h < 128; ++h) a += w[h]*xr[h];
      s[j] = a; m = fmaxf(m, a);
    }
    float sum = 0.0f;
    for (int j = 0; j < 9; ++j) { s[j] = __expf(s[j]-m); sum += s[j]; }
    float inv = 1.0f/sum;
    for (int j = 0; j < 9; ++j) out_x[t*9+j] = s[j]*inv;
  }
  { // y (9)
    float m = -1e30f;
    for (int j = 0; j < 9; ++j) {
      float a = by[j]; const float* w = Wy + j*128;
      for (int h = 0; h < 128; ++h) a += w[h]*xr[h];
      s[j] = a; m = fmaxf(m, a);
    }
    float sum = 0.0f;
    for (int j = 0; j < 9; ++j) { s[j] = __expf(s[j]-m); sum += s[j]; }
    float inv = 1.0f/sum;
    for (int j = 0; j < 9; ++j) out_y[t*9+j] = s[j]*inv;
  }
  { // attb[t][g] = att · b_type_g  (used by folded target score)
    const float* ar = att + (size_t)t*128;
    float a0=0,a1=0,a2=0,a3=0;
    for (int h = 0; h < 128; ++h) {
      float a = ar[h];
      a0 += a*bt0[h]; a1 += a*bt1[h]; a2 += a*bt2[h]; a3 += a*bt3[h];
    }
    attb[t*4+0]=a0; attb[t*4+1]=a1; attb[t*4+2]=a2; attb[t*4+3]=a3;
  }
}

// ====== target scores via fold: att·emb = (att@W_g)·basic + att·b_g ==========
__global__ void target_kernel(
    const float* __restrict__ ahu, const float* __restrict__ ehu,
    const float* __restrict__ anhu, const float* __restrict__ enhu,
    const float* __restrict__ Wb, const float* __restrict__ bb,
    const float* __restrict__ att2, const float* __restrict__ attb,
    float* __restrict__ tscore)
{
  int u = blockIdx.x*256 + threadIdx.x;
  if (u >= TOK*38) return;
  int t = u / 38, n = u % 38;
  int g, nn, Ng; const float* base;
  if (n < 1)       { g=0; nn=n;    Ng=1;  base=ahu;  }
  else if (n < 6)  { g=1; nn=n-1;  Ng=5;  base=ehu;  }
  else if (n < 22) { g=2; nn=n-6;  Ng=16; base=anhu; }
  else             { g=3; nn=n-22; Ng=16; base=enhu; }
  const float* uv = base + ((size_t)t*Ng + nn)*9;
  float u9[9];
  #pragma unroll
  for (int f = 0; f < 9; ++f) u9[f] = uv[f];
  const float* a2 = att2 + ((size_t)g*TOK + t)*128;
  float acc = attb[t*4 + g];
  for (int h = 0; h < 128; ++h) {
    const float* w = Wb + h*9;
    float bv = bb[h];
    #pragma unroll
    for (int f = 0; f < 9; ++f) bv += w[f]*u9[f];
    acc += a2[h] * fmaxf(bv, 0.0f);
  }
  tscore[u] = acc;
}

__global__ void tsoftmax_kernel(const float* __restrict__ tscore, float* __restrict__ out){
  int t = blockIdx.x*256 + threadIdx.x;
  if (t >= TOK) return;
  const float* s = tscore + (size_t)t*38;
  float m = -1e30f;
  for (int j = 0; j < 38; ++j) m = fmaxf(m, s[j]);
  float sum = 0.0f;
  for (int j = 0; j < 38; ++j) sum += __expf(s[j]-m);
  float inv = 1.0f/sum;
  for (int j = 0; j < 38; ++j) out[(size_t)t*38 + j] = __expf(s[j]-m)*inv;
}

// =============================================================================
extern "C" void kernel_launch(void* const* d_in, const int* in_sizes, int n_in,
                              void* d_out, int out_size, void* d_ws, size_t ws_size,
                              hipStream_t stream)
{
  (void)in_sizes; (void)n_in; (void)out_size; (void)ws_size;
  const float* env   = (const float*)d_in[0];
  const float* ahu   = (const float*)d_in[1];
  const float* ehu   = (const float*)d_in[2];
  const float* anhu  = (const float*)d_in[3];
  const float* enhu  = (const float*)d_in[4];
  const float* h0    = (const float*)d_in[5];
  const float* c0    = (const float*)d_in[6];
  const float* W_env = (const float*)d_in[7];  const float* b_env = (const float*)d_in[8];
  const float* W_bas = (const float*)d_in[9];  const float* b_bas = (const float*)d_in[10];
  const float* W_ah  = (const float*)d_in[11]; const float* b_ah  = (const float*)d_in[12];
  const float* W_eh  = (const float*)d_in[13]; const float* b_eh  = (const float*)d_in[14];
  const float* W_anh = (const float*)d_in[15]; const float* b_anh = (const float*)d_in[16];
  const float* W_enh = (const float*)d_in[17]; const float* b_enh = (const float*)d_in[18];
  const float* W_pre = (const float*)d_in[19]; const float* b_pre = (const float*)d_in[20];
  const float* W_ih  = (const float*)d_in[21]; const float* W_hh  = (const float*)d_in[22];
  const float* b_ih  = (const float*)d_in[23]; const float* b_hh  = (const float*)d_in[24];
  const float* W_enm = (const float*)d_in[25]; const float* b_enm = (const float*)d_in[26];
  const float* W_mx  = (const float*)d_in[27]; const float* b_mx  = (const float*)d_in[28];
  const float* W_my  = (const float*)d_in[29]; const float* b_my  = (const float*)d_in[30];
  const float* W_att = (const float*)d_in[31]; const float* b_att = (const float*)d_in[32];

  char* ws = (char*)d_ws;
  float*        x640   = (float*)(ws + 0);                    // 16384*640 f32
  float*        xpre   = (float*)(ws + 41943040);             // 16384*128 f32
  float*        gatesx = (float*)(ws + 50331648);             // 16384*512 f32
  float*        xlstm  = (float*)(ws + 83886080);             // 16384*128 f32
  float*        attbuf = (float*)(ws + 92274688);             // 16384*128 f32
  float*        att2   = (float*)(ws + 100663296);            // 4*16384*128 f32
  unsigned int* pooled = (unsigned int*)(ws + 100663296);     // aliases att2 (disjoint phases)
  float*        attb   = (float*)(ws + 134217728);            // 16384*4 f32
  float*        tscore = (float*)(ws + 134479872);            // 16384*38 f32
  float*        bsum   = (float*)(ws + 136970240);            // 512 f32

  float* out      = (float*)d_out;
  float* out_enum = out;                 // [B,S,3]
  float* out_x    = out + 49152;         // [B,S,9]
  float* out_y    = out + 196608;        // [B,S,9]
  float* out_t    = out + 344064;        // [B,S,38]
  float* out_hn   = out + 966656;        // [1,B,128]
  float* out_cn   = out + 974848;        // [1,B,128]

  // phase 0: init + bias sum
  init_pooled_kernel<<<(TOK*512 + 255)/256, 256, 0, stream>>>(pooled);
  bias_sum_kernel<<<2, 256, 0, stream>>>(b_ih, b_hh, bsum);

  // phase 1: fused unit embed + max-pool (WMMA)
  unit_embed_kernel<<<TOK*1/128,  256, 0, stream>>>(ahu,  W_bas, b_bas, W_ah,  b_ah,  pooled, 1,  0, TOK*1);
  unit_embed_kernel<<<TOK*5/128,  256, 0, stream>>>(ehu,  W_bas, b_bas, W_eh,  b_eh,  pooled, 5,  1, TOK*5);
  unit_embed_kernel<<<TOK*16/128, 256, 0, stream>>>(anhu, W_bas, b_bas, W_anh, b_anh, pooled, 16, 2, TOK*16);
  unit_embed_kernel<<<TOK*16/128, 256, 0, stream>>>(enhu, W_bas, b_bas, W_enh, b_enh, pooled, 16, 3, TOK*16);

  // phase 2: env embed + pooled decode -> x640
  env_kernel<<<(TOK*128 + 255)/256, 256, 0, stream>>>(env, W_env, b_env, x640);
  decode_pooled_kernel<<<(TOK*512 + 255)/256, 256, 0, stream>>>(pooled, x640);

  // phase 3: pre projection (relu), gate pre-activations (WMMA)
  dim3 gpre(TOK/128, 128/16);
  gemm_wmma_kernel<<<gpre, 256, 0, stream>>>(x640, W_pre, b_pre, xpre, TOK, 128, 640, 640, 640, 128, 1, 0);
  dim3 ggat(TOK/128, 512/16);
  gemm_wmma_kernel<<<ggat, 256, 0, stream>>>(xpre, W_ih, bsum, gatesx, TOK, 512, 128, 128, 128, 512, 0, 0);

  // phase 4: persistent WMMA LSTM (4 batch-tile blocks, no inter-block sync)
  lstm_kernel<<<4, 256, 0, stream>>>(gatesx, W_hh, h0, c0, xlstm, out_hn, out_cn);

  // phase 5: attention vector + folded att2_g = att @ W_g (WMMA, transB)
  dim3 gatt(TOK/128, 128/16);
  gemm_wmma_kernel<<<gatt, 256, 0, stream>>>(xlstm, W_att, b_att, attbuf, TOK, 128, 128, 128, 128, 128, 0, 0);
  const float* Wts[4] = {W_ah, W_eh, W_anh, W_enh};
  for (int g = 0; g < 4; ++g)
    gemm_wmma_kernel<<<gatt, 256, 0, stream>>>(attbuf, Wts[g], nullptr,
        att2 + (size_t)g*TOK*128, TOK, 128, 128, 128, 128, 128, 0, 1);

  // phase 6: heads + target
  heads_kernel<<<TOK/256, 256, 0, stream>>>(xlstm, attbuf, W_enm, b_enm, W_mx, b_mx,
      W_my, b_my, b_ah, b_eh, b_anh, b_enh, out_enum, out_x, out_y, attb);
  target_kernel<<<(TOK*38 + 255)/256, 256, 0, stream>>>(ahu, ehu, anhu, enhu,
      W_bas, b_bas, att2, attb, tscore);
  tsoftmax_kernel<<<TOK/256, 256, 0, stream>>>(tscore, out_t);
}
